// GRU_6631429505064
// MI455X (gfx1250) — matrix operand order repaired and checked by reading, not compiled, not run on hardware
//
#include <hip/hip_runtime.h>
#include <hip/hip_bf16.h>

// ---------------------------------------------------------------------------
// GRU scan on gfx1250 (MI455X).
//   B=32, L=2048, I=256, H=512, K = H+I = 768.
//   2 persistent workgroups x 16 batch rows; 1024 threads (32 waves) per WG,
//   one 16-col N-tile per wave. h lives in REGISTERS (8 per lane, owner-only
//   access per the WMMA D layout); only the bf16 copy is in LDS for the GEMMs.
//   Weights pre-packed fp32->bf16 into per-lane WMMA B-fragment layout.
//   x double-buffered in LDS via global_load_async_to_lds_b128 (ASYNCcnt).
//   3 barriers per step; gates use v_exp_f32 + v_rcp_f32 (no IEEE div).
// ---------------------------------------------------------------------------

typedef __bf16 bf16_t;
typedef __attribute__((ext_vector_type(16))) __bf16 v16bf;
typedef __attribute__((ext_vector_type(8)))  __bf16 v8bf;
typedef __attribute__((ext_vector_type(8)))  float  v8f;
typedef __attribute__((ext_vector_type(4)))  float  v4f;

static constexpr int Bsz  = 32;
static constexpr int Lsz  = 2048;
static constexpr int Isz  = 256;
static constexpr int Hsz  = 512;
static constexpr int Ksz  = Hsz + Isz;     // 768
static constexpr int KT   = Ksz / 32;      // 24 k-tiles of 32
static constexpr int NT   = Hsz / 16;      // 32 n-tiles of 16
static constexpr int PACKED_ELEMS = 3 * Hsz * Ksz;   // z, r, w packed bf16

__device__ __forceinline__ float fast_sigmoid(float x) {
    // 1/(1+e^-x) with v_rcp_f32 (no IEEE div expansion).
    return __builtin_amdgcn_rcpf(1.0f + __expf(-x));
}
__device__ __forceinline__ float fast_tanh(float x) {
    // 1 - 2/(e^{2x}+1); saturates correctly at +-inf.
    return 1.0f - 2.0f * __builtin_amdgcn_rcpf(__expf(2.0f * x) + 1.0f);
}

// Async copy 16 bytes: global -> LDS, tracked by ASYNCcnt (CDNA5).
__device__ __forceinline__ void async_b128_to_lds(unsigned lds_off,
                                                  const float* gsrc) {
    asm volatile("global_load_async_to_lds_b128 %0, %1, off"
                 :: "v"(lds_off), "v"(gsrc)
                 : "memory");
}
__device__ __forceinline__ void wait_asynccnt_le1() {
    // Asyncs complete in order: <=1 outstanding means the older buffer landed.
    asm volatile("s_wait_asynccnt 0x1" ::: "memory");
}

// ---------------------------------------------------------------------------
// Pack W[g] (fp32 [512,768] row-major) into bf16 WMMA B-fragments:
//   frag(g, nt, kt): 32 lanes x 16 contiguous bf16,
//   lane l -> n = nt*16 + l%16, k = kt*32 + (l/16)*8 + (j%8) + (j/8)*16 (j = 0..15).
// ---------------------------------------------------------------------------
__global__ void gru_pack_weights(const float* __restrict__ Wz,
                                 const float* __restrict__ Wr,
                                 const float* __restrict__ Ww,
                                 bf16_t* __restrict__ wp) {
    int p = blockIdx.x * blockDim.x + threadIdx.x;
    if (p >= PACKED_ELEMS) return;
    int j    = p & 15;
    int lane = (p >> 4) & 31;
    int q    = p >> 9;          // (g*NT + nt)*KT + kt
    int kt   = q % KT;
    int rem  = q / KT;          // g*NT + nt
    int nt   = rem & (NT - 1);
    int g    = rem >> 5;
    int n = nt * 16 + (lane & 15);
    int k = kt * 32 + (lane >> 4) * 8 + (j & 7) + (j >> 3) * 16;
    const float* W = (g == 0) ? Wz : ((g == 1) ? Wr : Ww);
    wp[p] = (bf16_t)W[n * Ksz + k];
}

// ---------------------------------------------------------------------------
// Persistent GRU scan: grid = 2 blocks, 1024 threads (32 waves).
// Block bb owns batch rows [bb*16, bb*16+16). Wave w owns n-tile w.
// ---------------------------------------------------------------------------
__global__ void __launch_bounds__(1024, 1)
gru_scan_kernel(const float* __restrict__ x,     // [B, L, I]
                const float* __restrict__ h0,    // [B, H]
                const bf16_t* __restrict__ wp,   // packed weights (3 gates)
                float* __restrict__ out) {       // [L, B, H]
    __shared__ bf16_t cat_zr[16 * Ksz];      // [h   | x_t] bf16     (24 KB)
    __shared__ bf16_t cat_w [16 * Ksz];      // [r*h | x_t] bf16     (24 KB)
    __shared__ float  xstage[2][16 * Isz];   // async x double buf   (32 KB)

    const int tid  = threadIdx.x;
    const int wave = tid >> 5;               // 0..31  == n-tile
    const int lane = tid & 31;
    const int half = lane >> 4;              // 0/1
    const int col  = lane & 15;
    const int nt   = wave;
    const int n    = nt * 16 + col;
    const int batch_base = blockIdx.x * 16;

    // Packed B-fragment base pointers (per gate, this wave's n-tile).
    const bf16_t* wz = wp + (size_t)((0 * NT + nt) * KT) * 512 + lane * 16;
    const bf16_t* wr = wp + (size_t)((1 * NT + nt) * KT) * 512 + lane * 16;
    const bf16_t* ww = wp + (size_t)((2 * NT + nt) * KT) * 512 + lane * 16;

    // A-fragment per-lane LDS bases (16-bit A 16x32 layout):
    //   row m = lane%16; chunks at k = kt*32 + half*8 and +16.
    const bf16_t* arow_zr = &cat_zr[col * Ksz + half * 8];
    const bf16_t* arow_w  = &cat_w [col * Ksz + half * 8];

    // x staging ownership: thread -> (row = tid/64, 4 floats at (tid%64)*4).
    const int xrow = tid >> 6;
    const int xi0  = (tid & 63) * 4;
    const float* xsrc_base =
        x + (size_t)(batch_base + xrow) * Lsz * Isz + xi0;
    const unsigned xstage_off0 =
        (unsigned)(uintptr_t)&xstage[0][xrow * Isz + xi0];
    const unsigned xstage_off1 =
        (unsigned)(uintptr_t)&xstage[1][xrow * Isz + xi0];

    // --- init h: fp32 master in registers (owner-lane cells), bf16 in LDS ---
    // D layout: vgpr v, lane -> (m = v + 8*half, n). h[m][n] is only ever
    // touched by its owner lane, so it never needs to round-trip LDS.
    float hreg[8];
    #pragma unroll
    for (int v = 0; v < 8; ++v) {
        int m = v + 8 * half;
        float hv = h0[(size_t)(batch_base + m) * Hsz + n];
        hreg[v] = hv;
        cat_zr[m * Ksz + n] = (bf16_t)hv;
    }
    // Kick off async copy of x_0 into buffer 0.
    async_b128_to_lds(xstage_off0, xsrc_base);

    float zv[8];

    for (int t = 0; t < Lsz; ++t) {
        // ---- x_t: issue prefetch of x_{t+1}, wait for x_t, convert ---------
        {
            int tn = (t + 1 < Lsz) ? (t + 1) : t;   // clamped prefetch
            async_b128_to_lds((t & 1) ? xstage_off0 : xstage_off1,
                              xsrc_base + (size_t)tn * Isz);
            wait_asynccnt_le1();                    // oldest (x_t) landed
            const float* sp = &xstage[t & 1][xrow * Isz + xi0];
            v4f a = *(const v4f*)sp;
            bf16_t* c0 = &cat_zr[xrow * Ksz + Hsz + xi0];
            bf16_t* c1 = &cat_w [xrow * Ksz + Hsz + xi0];
            c0[0] = (bf16_t)a.x; c0[1] = (bf16_t)a.y;
            c0[2] = (bf16_t)a.z; c0[3] = (bf16_t)a.w;
            c1[0] = (bf16_t)a.x; c1[1] = (bf16_t)a.y;
            c1[2] = (bf16_t)a.z; c1[3] = (bf16_t)a.w;
        }
        __syncthreads();   // [A] x_t staged + h from step t-1 visible

        // ---- z and r GEMMs over full K (shared A-fragment) -----------------
        v8f az = {}, ar = {};
        #pragma unroll 4
        for (int kt = 0; kt < KT; ++kt) {
            v8bf a_lo = *(const v8bf*)(arow_zr + kt * 32);
            v8bf a_hi = *(const v8bf*)(arow_zr + kt * 32 + 16);
            v16bf a = __builtin_shufflevector(a_lo, a_hi,
                0, 1, 2, 3, 4, 5, 6, 7, 8, 9, 10, 11, 12, 13, 14, 15);
            v16bf bz = *(const v16bf*)(wz + kt * 512);
            v16bf br = *(const v16bf*)(wr + kt * 512);
            az = __builtin_amdgcn_wmma_f32_16x16x32_bf16(false, a, false, bz,
                                                         (short)0, az, false, false);
            ar = __builtin_amdgcn_wmma_f32_16x16x32_bf16(false, a, false, br,
                                                         (short)0, ar, false, false);
        }

        // Gates + write r*h (bf16) into cat_w's h-part (owner cells only).
        #pragma unroll
        for (int v = 0; v < 8; ++v) {
            int m = v + 8 * half;
            zv[v] = fast_sigmoid(az[v]);
            float r = fast_sigmoid(ar[v]);
            cat_w[m * Ksz + n] = (bf16_t)(r * hreg[v]);
        }
        __syncthreads();   // [B] r*h staged in cat_w

        // ---- candidate GEMM: tanh([r*h, x] @ Ww^T) -------------------------
        v8f aw = {};
        #pragma unroll 4
        for (int kt = 0; kt < KT; ++kt) {
            v8bf a_lo = *(const v8bf*)(arow_w + kt * 32);
            v8bf a_hi = *(const v8bf*)(arow_w + kt * 32 + 16);
            v16bf a = __builtin_shufflevector(a_lo, a_hi,
                0, 1, 2, 3, 4, 5, 6, 7, 8, 9, 10, 11, 12, 13, 14, 15);
            v16bf bw = *(const v16bf*)(ww + kt * 512);
            aw = __builtin_amdgcn_wmma_f32_16x16x32_bf16(false, a, false, bw,
                                                         (short)0, aw, false, false);
        }

        // ---- h_new = h + z*(tanh(.) - h); update hreg, cat_zr, out ---------
        float* outt = out + (size_t)t * Bsz * Hsz + (size_t)batch_base * Hsz;
        #pragma unroll
        for (int v = 0; v < 8; ++v) {
            int m = v + 8 * half;
            float hh = fast_tanh(aw[v]);
            float ho = hreg[v];
            float hn = ho + zv[v] * (hh - ho);
            hreg[v] = hn;
            cat_zr[m * Ksz + n] = (bf16_t)hn;
            outt[m * Hsz + n] = hn;
        }
        __syncthreads();   // [E] step fully retired before next x staging
    }
}

extern "C" void kernel_launch(void* const* d_in, const int* in_sizes, int n_in,
                              void* d_out, int out_size, void* d_ws, size_t ws_size,
                              hipStream_t stream) {
    const float* x  = (const float*)d_in[0];   // [32, 2048, 256]
    const float* h0 = (const float*)d_in[1];   // [32, 512]
    const float* Wz = (const float*)d_in[2];   // [512, 768]
    const float* Wr = (const float*)d_in[3];
    const float* Ww = (const float*)d_in[4];
    float* out = (float*)d_out;                // [2048, 32, 512]
    bf16_t* wp = (bf16_t*)d_ws;                // 2.25 MB packed weights

    (void)in_sizes; (void)n_in; (void)out_size; (void)ws_size;

    {
        int threads = 256;
        int blocks = (PACKED_ELEMS + threads - 1) / threads;
        gru_pack_weights<<<blocks, threads, 0, stream>>>(Wz, Wr, Ww, wp);
    }
    gru_scan_kernel<<<dim3(2), dim3(1024), 0, stream>>>(x, h0, wp, out);
}
